// DGCNNEncOld_7705171329414
// MI455X (gfx1250) — compile-verified, matrix operand order checked by reading, not run on hardware
//
#include <hip/hip_runtime.h>

typedef __attribute__((ext_vector_type(16))) __bf16 v16bf;
typedef __attribute__((ext_vector_type(8)))  float  v8f;

#define LEAKY 0.2f
#define BN_EPS 1e-5f
#define FBIG 3.402823e38f

// ---------- CDNA5 async-to-LDS probe (fallback: plain stores) ----------
#if defined(__has_builtin)
#if __has_builtin(__builtin_amdgcn_global_load_async_to_lds_b128)
#define HAVE_ASYNC_LDS 1
#endif
#endif

static __device__ inline void wait_async0() {
#if defined(__has_builtin)
#if __has_builtin(__builtin_amdgcn_s_wait_asynccnt)
  __builtin_amdgcn_s_wait_asynccnt(0);
#else
  asm volatile("s_wait_asynccnt 0" ::: "memory");
#endif
#else
  asm volatile("s_wait_asynccnt 0" ::: "memory");
#endif
}

// copy 8 bf16 (16B) global -> LDS, async when available
static __device__ inline void cp_g2l_16B(__bf16* ldst, const __bf16* gsrc) {
#ifdef HAVE_ASYNC_LDS
  // param 0: int4 addrspace(1)* (printed as 'int4 __device__*'), non-const
  // param 1: int4 addrspace(3)*
  typedef __attribute__((ext_vector_type(4))) int i4;
  typedef __attribute__((address_space(1))) i4 gi4;
  typedef __attribute__((address_space(3))) i4 li4;
  __builtin_amdgcn_global_load_async_to_lds_b128(
      (gi4*)(unsigned long long)(const void*)gsrc,
      (li4*)(unsigned)(unsigned long long)(const void*)ldst, 0, 0);
#else
  *(uint4*)ldst = *(const uint4*)gsrc;
#endif
}

// ---------- helpers ----------

static __device__ inline v16bf ld_frag16(const __bf16* p) {
  // 16 bf16 = 32 bytes, loaded as two 128-bit chunks (K-halves per ISA layout)
  union { uint4 u[2]; v16bf v; } t;
  t.u[0] = *(const uint4*)(p);
  t.u[1] = *(const uint4*)(p + 16);
  return t.v;
}

// ---------- data prep kernels ----------

// f32 rows [R,d] -> bf16 padded rows [R,dpad] + squared norms
__global__ void k_rows_bf16(const float* __restrict__ X, int d,
                            __bf16* __restrict__ Xb, int dpad,
                            float* __restrict__ sq, int R) {
  int r = blockIdx.x * blockDim.x + threadIdx.x;
  if (r >= R) return;
  float acc = 0.f;
  for (int j = 0; j < dpad; ++j) {
    float v = (j < d) ? X[(size_t)r * d + j] : 0.f;
    Xb[(size_t)r * dpad + j] = (__bf16)v;
    acc += v * v;
  }
  sq[r] = acc;
}

// W [Kd,Nd] f32 -> Wt [Nd,Kp] bf16 (transposed, K-padded) for B fragments
__global__ void k_pack_w(const float* __restrict__ W, int Kd, int Nd,
                         __bf16* __restrict__ Wt, int Kp) {
  int t = blockIdx.x * blockDim.x + threadIdx.x;
  if (t >= Nd * Kp) return;
  int nn = t / Kp, kk = t % Kp;
  float v = (kk < Kd) ? W[(size_t)kk * Nd + nn] : 0.f;
  Wt[(size_t)nn * Kp + kk] = (__bf16)v;
}

// ---------- fused kNN: WMMA Gram + distance strip in LDS + top-k ----------
// grid: (n/16, B), block: 256 threads (8 waves). n must be 4096.
// KT = number of 32-wide K tiles (compile-time so fragments stay in fixed regs)
template <int KT>
__global__ void k_knn(const __bf16* __restrict__ xbf, const float* __restrict__ sq,
                      int* __restrict__ idx, int n, int knn) {
  __shared__ float dist[16 * 4096];            // 256 KB strip, fits 320 KB WGP LDS
  const int dpad = KT * 32;
  const int b   = blockIdx.y;
  const int m0  = blockIdx.x * 16;
  const __bf16* xb  = xbf + (size_t)b * n * dpad;
  const float*  sqb = sq  + (size_t)b * n;
  const int lane = threadIdx.x & 31;
  const int wave = threadIdx.x >> 5;
  const int half = lane >> 4, lr = lane & 15;

  // preload A fragments (the 16 query rows) once per k-tile
  v16bf afr[KT];
#pragma unroll
  for (int kt = 0; kt < KT; ++kt)
    afr[kt] = ld_frag16(xb + (size_t)(m0 + lr) * dpad + kt * 32 + half * 8);

  const int ntiles = n >> 4;
  for (int ct = wave; ct < ntiles; ct += 8) {
    const int c0 = ct * 16;
    v8f acc = {};
#pragma unroll
    for (int kt = 0; kt < KT; ++kt) {
      v16bf bfr = ld_frag16(xb + (size_t)(c0 + lr) * dpad + kt * 32 + half * 8);
      acc = __builtin_amdgcn_wmma_f32_16x16x32_bf16(
          false, afr[kt], false, bfr, (short)0, acc, false, false);
    }
    const float scol = sqb[c0 + lr];
#pragma unroll
    for (int i = 0; i < 8; ++i) {
      int row = half * 8 + i;                  // C layout: VGPR i, lane half
      dist[row * 4096 + c0 + lr] = sqb[m0 + row] + scol - 2.0f * acc[i];
    }
  }
  __syncthreads();

  // top-k smallest per row: 8 waves x 2 rows, iterative wave argmin
  for (int rr = 0; rr < 2; ++rr) {
    const int row = wave * 2 + rr;
    float* drow = &dist[row * 4096];
    for (int sel = 0; sel < knn; ++sel) {
      float mv = FBIG; int mi = 0;
      for (int c = lane; c < n; c += 32) {
        float v = drow[c];
        if (v < mv) { mv = v; mi = c; }
      }
#pragma unroll
      for (int off = 16; off > 0; off >>= 1) {
        float ov = __shfl_down(mv, off, 32);
        int   oi = __shfl_down(mi, off, 32);
        if (ov < mv) { mv = ov; mi = oi; }
      }
      mi = __shfl(mi, 0, 32);
      if (lane == 0) {
        idx[(size_t)(b * n + m0 + row) * knn + sel] = b * n + mi;
        drow[mi] = FBIG;
      }
    }
  }
}

// ---------- edge feature gather: [xi | xj - xi] bf16, K-padded ----------
__global__ void k_gather_edge(const float* __restrict__ X, int d,
                              const int* __restrict__ idx, int knn,
                              __bf16* __restrict__ Eb, int ld2, int total) {
  int t = blockIdx.x * blockDim.x + threadIdx.x;
  if (t >= total) return;
  int f  = t % ld2;
  int er = t / ld2;
  int r  = er / knn, kk = er % knn;
  float v = 0.f;
  if (f < d) {
    v = X[(size_t)r * d + f];
  } else if (f < 2 * d) {
    int j = idx[(size_t)r * knn + kk];
    int c = f - d;
    v = X[(size_t)j * d + c] - X[(size_t)r * d + c];
  }
  Eb[(size_t)er * ld2 + f] = (__bf16)v;
}

// ---------- generic WMMA GEMM: C[M,N] = A[M,KP] * Bt[N,KP]^T ----------
// KP: compile-time K depth (multiple of 32). NT: output tiles along N per wave.
// All 4 waves of a block share the same N-tile strip, so the B tile
// (NT*16 rows x KP bf16) is staged once into LDS (async-to-LDS when the
// toolchain exposes it), then each wave reads fragments via ds_load_b128.
// grid: (Mtiles/4, N/(16*NT)), block 128 (4 waves)
template <int KP, int NT>
__global__ void k_gemm(const __bf16* __restrict__ A, const __bf16* __restrict__ Bt,
                       float* __restrict__ C, int N, int Mtiles) {
  __shared__ __bf16 Bs[NT * 16 * KP];          // <= 24 KB
  const int lane = threadIdx.x & 31;
  const int wave = threadIdx.x >> 5;
  const int mt  = blockIdx.x * 4 + wave;
  const int nt0 = blockIdx.y * NT;

  // cooperative stage of the B tile (16-byte chunks)
  const int chunks = (NT * 16 * KP) / 8;
  for (int cc = threadIdx.x; cc < chunks; cc += 128)
    cp_g2l_16B(Bs + cc * 8, Bt + (size_t)nt0 * 16 * KP + cc * 8);
#ifdef HAVE_ASYNC_LDS
  wait_async0();
#endif
  __syncthreads();

  if (mt >= Mtiles) return;
  const int half = lane >> 4, lr = lane & 15;
  const __bf16* arow = A + (size_t)(mt * 16 + lr) * KP + half * 8;
  v8f acc[NT];
#pragma unroll
  for (int j = 0; j < NT; ++j) acc[j] = (v8f){};
#pragma unroll
  for (int kt = 0; kt < KP / 32; ++kt) {
    if (kt + 1 < KP / 32)
      __builtin_prefetch(arow + kt * 32 + 32, 0, 1);   // global_prefetch_b8
    v16bf a = ld_frag16(arow + kt * 32);
#pragma unroll
    for (int j = 0; j < NT; ++j) {
      v16bf bb = ld_frag16(Bs + (j * 16 + lr) * KP + kt * 32 + half * 8);
      acc[j] = __builtin_amdgcn_wmma_f32_16x16x32_bf16(
          false, a, false, bb, (short)0, acc[j], false, false);
    }
  }
#pragma unroll
  for (int j = 0; j < NT; ++j) {
    float* crow = C + (size_t)(mt * 16 + half * 8) * N + (nt0 + j) * 16 + lr;
#pragma unroll
    for (int i = 0; i < 8; ++i) crow[(size_t)i * N] = acc[j][i];
  }
}

// ---------- BatchNorm column stats (sum/sumsq via f32 atomics) ----------
__global__ void k_colstats(const float* __restrict__ X, int rows, int cols, int rchunk,
                           float* __restrict__ sum, float* __restrict__ sumsq) {
  int c = blockIdx.y * blockDim.x + threadIdx.x;
  if (c >= cols) return;
  int r0 = blockIdx.x * rchunk;
  int r1 = min(rows, r0 + rchunk);
  float s = 0.f, s2 = 0.f;
  for (int r = r0; r < r1; ++r) {
    float v = X[(size_t)r * cols + c];
    s += v; s2 += v * v;
  }
  atomicAdd(&sum[c], s);
  atomicAdd(&sumsq[c], s2);
}

// ---------- BN + LeakyReLU ----------
__global__ void k_bn_act_f32(float* __restrict__ X, int total, int cols,
                             const float* __restrict__ sum, const float* __restrict__ sumsq,
                             const float* __restrict__ g, const float* __restrict__ b,
                             float invN) {
  int t = blockIdx.x * blockDim.x + threadIdx.x;
  if (t >= total) return;
  int c = t % cols;
  float mean = sum[c] * invN;
  float var  = sumsq[c] * invN - mean * mean;
  float y = g[c] * (X[t] - mean) * rsqrtf(var + BN_EPS) + b[c];
  X[t] = (y > 0.f) ? y : LEAKY * y;
}

__global__ void k_bn_act_bf16(const float* __restrict__ X, int total, int cols,
                              const float* __restrict__ sum, const float* __restrict__ sumsq,
                              const float* __restrict__ g, const float* __restrict__ b,
                              float invN, __bf16* __restrict__ Y) {
  int t = blockIdx.x * blockDim.x + threadIdx.x;
  if (t >= total) return;
  int c = t % cols;
  float mean = sum[c] * invN;
  float var  = sumsq[c] * invN - mean * mean;
  float y = g[c] * (X[t] - mean) * rsqrtf(var + BN_EPS) + b[c];
  y = (y > 0.f) ? y : LEAKY * y;
  Y[t] = (__bf16)y;
}

// ---------- max over k neighbors ----------
__global__ void k_maxk(const float* __restrict__ A, int knn, int cols,
                       float* __restrict__ O, int total) {
  int t = blockIdx.x * blockDim.x + threadIdx.x;
  if (t >= total) return;
  int r = t / cols, c = t % cols;
  float m = -FBIG;
  for (int kk = 0; kk < knn; ++kk)
    m = fmaxf(m, A[(size_t)(r * knn + kk) * cols + c]);
  O[t] = m;
}

// ---------- concat x1|x2|x3 -> bf16 [R,192] ----------
__global__ void k_concat(const float* __restrict__ x1, const float* __restrict__ x2,
                         const float* __restrict__ x3, __bf16* __restrict__ F, int total) {
  int t = blockIdx.x * blockDim.x + threadIdx.x;
  if (t >= total) return;
  int r = t / 192, c = t % 192;
  float v = (c < 64) ? x1[(size_t)r * 64 + c]
          : (c < 128) ? x2[(size_t)r * 64 + (c - 64)]
                      : x3[(size_t)r * 64 + (c - 128)];
  F[t] = (__bf16)v;
}

// ---------- global max over N per batch ----------
__global__ void k_maxn(const float* __restrict__ H, int n, int cols,
                       float* __restrict__ G) {
  int b = blockIdx.x;
  int c = blockIdx.y * blockDim.x + threadIdx.x;
  if (c >= cols) return;
  float m = -FBIG;
  for (int i = 0; i < n; ++i)
    m = fmaxf(m, H[(size_t)(b * n + i) * cols + c]);
  G[(size_t)b * cols + c] = m;
}

// ---------- broadcast global encoding ----------
__global__ void k_bcast(const float* __restrict__ G, float* __restrict__ O,
                        int n, int cols, int total) {
  int t = blockIdx.x * blockDim.x + threadIdx.x;
  if (t >= total) return;
  int r = t / cols, c = t % cols;
  O[t] = G[(size_t)(r / n) * cols + c];
}

// ---------- host orchestration ----------

extern "C" void kernel_launch(void* const* d_in, const int* in_sizes, int n_in,
                              void* d_out, int out_size, void* d_ws, size_t ws_size,
                              hipStream_t stream) {
  const float* w1a = (const float*)d_in[2];
  const float* g1a = (const float*)d_in[3];
  const float* b1a = (const float*)d_in[4];
  const float* w1b = (const float*)d_in[5];
  const float* g1b = (const float*)d_in[6];
  const float* b1b = (const float*)d_in[7];
  const float* w2a = (const float*)d_in[8];
  const float* g2a = (const float*)d_in[9];
  const float* b2a = (const float*)d_in[10];
  const float* w2b = (const float*)d_in[11];
  const float* g2b = (const float*)d_in[12];
  const float* b2b = (const float*)d_in[13];
  const float* w3a = (const float*)d_in[14];
  const float* g3a = (const float*)d_in[15];
  const float* b3a = (const float*)d_in[16];
  const float* wm  = (const float*)d_in[17];
  const float* gm  = (const float*)d_in[18];
  const float* bm  = (const float*)d_in[19];
  const float* x   = (const float*)d_in[20];

  const int nb  = in_sizes[1];          // 4
  const int R   = in_sizes[0] / 3;      // 16384
  const int n   = R / nb;               // 4096
  const int KNN = 20;
  const int E   = R * KNN;              // 327680
  const int EMB = 1024;

  // workspace bump allocator (256B aligned)
  char* wp = (char*)d_ws;
  auto walloc = [&](size_t bytes) -> void* {
    void* r = (void*)wp;
    wp += (bytes + 255) & ~(size_t)255;
    return r;
  };
  __bf16* xbf  = (__bf16*)walloc((size_t)R * 64 * sizeof(__bf16));
  float*  sq   = (float*)walloc((size_t)R * sizeof(float));
  int*    idxb = (int*)walloc((size_t)R * KNN * sizeof(int));
  __bf16* wtA  = (__bf16*)walloc((size_t)64 * 128 * sizeof(__bf16));
  __bf16* wtB  = (__bf16*)walloc((size_t)64 * 64 * sizeof(__bf16));
  __bf16* wtM  = (__bf16*)walloc((size_t)EMB * 192 * sizeof(__bf16));
  float*  stats= (float*)walloc((size_t)2 * EMB * sizeof(float));
  __bf16* ebf  = (__bf16*)walloc((size_t)E * 128 * sizeof(__bf16));
  float*  lin  = (float*)walloc((size_t)E * 64 * sizeof(float));
  __bf16* abf  = (__bf16*)walloc((size_t)E * 64 * sizeof(__bf16));
  __bf16* fbf  = (__bf16*)walloc((size_t)R * 192 * sizeof(__bf16));
  float*  hg   = (float*)walloc((size_t)R * EMB * sizeof(float));
  float*  xg   = (float*)walloc((size_t)nb * EMB * sizeof(float));

  float* x1 = (float*)d_out;
  float* x2 = x1 + (size_t)R * 64;
  float* x3 = x2 + (size_t)R * 64;
  float* ge = x3 + (size_t)R * 64;

  const float invE = 1.0f / (float)E;
  const int Mt = E / 16;                // 20480 edge-row tiles

  // ---- conv1: d=3 (Gram Kpad 32, MLP K 32) ----
  k_pack_w<<<dim3((64 * 32 + 255) / 256), 256, 0, stream>>>(w1a, 6, 64, wtA, 32);
  k_pack_w<<<dim3((64 * 64 + 255) / 256), 256, 0, stream>>>(w1b, 64, 64, wtB, 64);
  k_rows_bf16<<<dim3((R + 255) / 256), 256, 0, stream>>>(x, 3, xbf, 32, sq, R);
  k_knn<1><<<dim3(n / 16, nb), 256, 0, stream>>>(xbf, sq, idxb, n, KNN);
  k_gather_edge<<<dim3((E * 32 + 255) / 256), 256, 0, stream>>>(x, 3, idxb, KNN, ebf, 32, E * 32);
  k_gemm<32, 4><<<dim3(Mt / 4, 1), 128, 0, stream>>>(ebf, wtA, lin, 64, Mt);
  (void)hipMemsetAsync(stats, 0, 2 * 64 * sizeof(float), stream);
  k_colstats<<<dim3((E + 511) / 512, 1), 64, 0, stream>>>(lin, E, 64, 512, stats, stats + 64);
  k_bn_act_bf16<<<dim3((E * 64 + 255) / 256), 256, 0, stream>>>(
      lin, E * 64, 64, stats, stats + 64, g1a, b1a, invE, abf);
  k_gemm<64, 4><<<dim3(Mt / 4, 1), 128, 0, stream>>>(abf, wtB, lin, 64, Mt);
  (void)hipMemsetAsync(stats, 0, 2 * 64 * sizeof(float), stream);
  k_colstats<<<dim3((E + 511) / 512, 1), 64, 0, stream>>>(lin, E, 64, 512, stats, stats + 64);
  k_bn_act_f32<<<dim3((E * 64 + 255) / 256), 256, 0, stream>>>(
      lin, E * 64, 64, stats, stats + 64, g1b, b1b, invE);
  k_maxk<<<dim3((R * 64 + 255) / 256), 256, 0, stream>>>(lin, KNN, 64, x1, R * 64);

  // ---- conv2: d=64 (Gram Kpad 64, MLP K 128) ----
  k_pack_w<<<dim3((64 * 128 + 255) / 256), 256, 0, stream>>>(w2a, 128, 64, wtA, 128);
  k_pack_w<<<dim3((64 * 64 + 255) / 256), 256, 0, stream>>>(w2b, 64, 64, wtB, 64);
  k_rows_bf16<<<dim3((R + 255) / 256), 256, 0, stream>>>(x1, 64, xbf, 64, sq, R);
  k_knn<2><<<dim3(n / 16, nb), 256, 0, stream>>>(xbf, sq, idxb, n, KNN);
  k_gather_edge<<<dim3((E * 128 + 255) / 256), 256, 0, stream>>>(x1, 64, idxb, KNN, ebf, 128, E * 128);
  k_gemm<128, 4><<<dim3(Mt / 4, 1), 128, 0, stream>>>(ebf, wtA, lin, 64, Mt);
  (void)hipMemsetAsync(stats, 0, 2 * 64 * sizeof(float), stream);
  k_colstats<<<dim3((E + 511) / 512, 1), 64, 0, stream>>>(lin, E, 64, 512, stats, stats + 64);
  k_bn_act_bf16<<<dim3((E * 64 + 255) / 256), 256, 0, stream>>>(
      lin, E * 64, 64, stats, stats + 64, g2a, b2a, invE, abf);
  k_gemm<64, 4><<<dim3(Mt / 4, 1), 128, 0, stream>>>(abf, wtB, lin, 64, Mt);
  (void)hipMemsetAsync(stats, 0, 2 * 64 * sizeof(float), stream);
  k_colstats<<<dim3((E + 511) / 512, 1), 64, 0, stream>>>(lin, E, 64, 512, stats, stats + 64);
  k_bn_act_f32<<<dim3((E * 64 + 255) / 256), 256, 0, stream>>>(
      lin, E * 64, 64, stats, stats + 64, g2b, b2b, invE);
  k_maxk<<<dim3((R * 64 + 255) / 256), 256, 0, stream>>>(lin, KNN, 64, x2, R * 64);

  // ---- conv3: d=64, single layer ----
  k_pack_w<<<dim3((64 * 128 + 255) / 256), 256, 0, stream>>>(w3a, 128, 64, wtA, 128);
  k_rows_bf16<<<dim3((R + 255) / 256), 256, 0, stream>>>(x2, 64, xbf, 64, sq, R);
  k_knn<2><<<dim3(n / 16, nb), 256, 0, stream>>>(xbf, sq, idxb, n, KNN);
  k_gather_edge<<<dim3((E * 128 + 255) / 256), 256, 0, stream>>>(x2, 64, idxb, KNN, ebf, 128, E * 128);
  k_gemm<128, 4><<<dim3(Mt / 4, 1), 128, 0, stream>>>(ebf, wtA, lin, 64, Mt);
  (void)hipMemsetAsync(stats, 0, 2 * 64 * sizeof(float), stream);
  k_colstats<<<dim3((E + 511) / 512, 1), 64, 0, stream>>>(lin, E, 64, 512, stats, stats + 64);
  k_bn_act_f32<<<dim3((E * 64 + 255) / 256), 256, 0, stream>>>(
      lin, E * 64, 64, stats, stats + 64, g3a, b3a, invE);
  k_maxk<<<dim3((R * 64 + 255) / 256), 256, 0, stream>>>(lin, KNN, 64, x3, R * 64);

  // ---- final projection 192 -> 1024, BN+leaky, global max, broadcast ----
  k_concat<<<dim3((R * 192 + 255) / 256), 256, 0, stream>>>(x1, x2, x3, fbf, R * 192);
  k_pack_w<<<dim3((EMB * 192 + 255) / 256), 256, 0, stream>>>(wm, 192, EMB, wtM, 192);
  k_gemm<192, 4><<<dim3(R / 16 / 4, EMB / 64), 128, 0, stream>>>(fbf, wtM, hg, EMB, R / 16);
  (void)hipMemsetAsync(stats, 0, 2 * EMB * sizeof(float), stream);
  k_colstats<<<dim3((R + 511) / 512, EMB / 64), 64, 0, stream>>>(hg, R, EMB, 512, stats, stats + EMB);
  k_bn_act_f32<<<dim3((R * EMB + 255) / 256), 256, 0, stream>>>(
      hg, R * EMB, EMB, stats, stats + EMB, gm, bm, 1.0f / (float)R);
  k_maxn<<<dim3(nb, EMB / 64), 64, 0, stream>>>(hg, n, EMB, xg);
  k_bcast<<<dim3((R * EMB + 255) / 256), 256, 0, stream>>>(xg, ge, n, EMB, R * EMB);
}
